// Gaussian_Fitting_21328807592200
// MI455X (gfx1250) — matrix operand order repaired and checked by reading, not compiled, not run on hardware
//
#include <hip/hip_runtime.h>

// MI455X / gfx1250, wave32. Fused Gaussian-kernel attention block.
//   S=128 seq, D=256 model, H=8 heads, HD=32 head_dim.
// All tensors are L2-resident; dominant cost is the pairwise-Gaussian
// transcendental loop (exploits K(s,t)=K(t,s) symmetry, 2^-d^2 via
// v_exp_f32 with pre-scaled operands -> 3 VALU + 1 TRANS per pair).
// GEMMs use v_wmma_f32_16x16x32_f16 with f32 accumulation.

typedef __attribute__((ext_vector_type(16))) _Float16 v16h;
typedef __attribute__((ext_vector_type(8)))  float    v8f;

#define SEQ    128
#define DMODEL 256
#define NHEADS 8
#define HDIM   32
#define LN_EPS 1e-5f
// sqrt(0.5 * log2(e)): exp(-0.5*u^2) == exp2(-(C_EXP2*u)^2)
#define C_EXP2 0.84932180028801904272f

static __device__ __forceinline__ v8f wmma_f16(v16h a, v16h b, v8f c) {
  return __builtin_amdgcn_wmma_f32_16x16x32_f16(
      /*neg_a=*/false, a, /*neg_b=*/false, b,
      /*c_mod=*/(short)0, c, /*reuse_a=*/false, /*reuse_b=*/false);
}

// ---------------------------------------------------------------------------
// Kernel 1: X = inputs @ w_in^T + b_in, LayerNorm over each head's 32 cols.
// One wave per (16-row tile, head); two WMMA accumulators span the head's
// HD=32 columns so the LN reduction is a 4-step 16-lane shfl butterfly.
// ---------------------------------------------------------------------------
__global__ __launch_bounds__(32)
void k_in_proj_ln(const float* __restrict__ inp,  const float* __restrict__ w_in,
                  const float* __restrict__ b_in, const float* __restrict__ ln_g,
                  const float* __restrict__ ln_b, float* __restrict__ Xn,
                  _Float16* __restrict__ Xh) {
  const int lane  = threadIdx.x;
  const int mTile = blockIdx.x;          // 0..7  (S/16)
  const int head  = blockIdx.y;          // 0..7
  const int nBase = head * HDIM;
  const int mlo = lane & 15, hi = lane >> 4;

  v8f c0 = {}; v8f c1 = {};
  for (int kb = 0; kb < DMODEL / 32; ++kb) {
    v16h a, b0, b1;
    const int mg = mTile * 16 + mlo;
#pragma unroll
    for (int h = 0; h < 16; ++h) {
      const int kA = kb * 32 + hi * 16 + h;                     // A-frag K
      a[h] = (_Float16)inp[mg * DMODEL + kA];
      const int kB = kb * 32 + hi * 16 + h;                     // B-frag K
      b0[h] = (_Float16)w_in[(nBase +      mlo) * DMODEL + kB]; // B = w_in^T
      b1[h] = (_Float16)w_in[(nBase + 16 + mlo) * DMODEL + kB];
    }
    c0 = wmma_f16(a, b0, c0);
    c1 = wmma_f16(a, b1, c1);
  }

  const float bias0 = b_in[nBase + mlo],  bias1 = b_in[nBase + 16 + mlo];
  const float g0 = ln_g[mlo], g1 = ln_g[16 + mlo];
  const float e0 = ln_b[mlo], e1 = ln_b[16 + mlo];

#pragma unroll
  for (int r = 0; r < 8; ++r) {
    float x0 = c0[r] + bias0, x1 = c1[r] + bias1;
    float s1 = x0 + x1, s2 = x0 * x0 + x1 * x1;
#pragma unroll
    for (int m = 1; m < 16; m <<= 1) {     // stays inside 16-lane half-wave
      s1 += __shfl_xor(s1, m, 32);
      s2 += __shfl_xor(s2, m, 32);
    }
    const float mu   = s1 * (1.0f / HDIM);
    const float var  = s2 * (1.0f / HDIM) - mu * mu;
    const float rstd = rsqrtf(var + LN_EPS);
    const float y0 = (x0 - mu) * rstd * g0 + e0;
    const float y1 = (x1 - mu) * rstd * g1 + e1;
    const int mg = mTile * 16 + r + 8 * hi;      // C-frag row mapping
    Xn[mg * DMODEL + nBase +      mlo] = y0;
    Xn[mg * DMODEL + nBase + 16 + mlo] = y1;
    Xh[mg * DMODEL + nBase +      mlo] = (_Float16)y0;
    Xh[mg * DMODEL + nBase + 16 + mlo] = (_Float16)y1;
  }
}

// ---------------------------------------------------------------------------
// Kernel 2: Gaussian logits L[s,t] = (1/HD) sum_{e,f} exp(-(x_se-x_tf)^2/2).
// L is symmetric: compute only upper-triangle 16x16 tiles (36 of 64) and
// mirror. Block = (triangle tile, head), 256 threads, one (s,t) per thread.
// Operands pre-scaled by C_EXP2 so body is d=a-b; acc+=exp2(-(d*d)).
// ---------------------------------------------------------------------------
__global__ __launch_bounds__(256)
void k_gauss_logits(const float* __restrict__ Xn, float* __restrict__ attnL) {
  __shared__ float xa[16][HDIM + 1];    // +1 pad: distinct LDS banks per row
  __shared__ float xb[16][HDIM + 1];

  // linear tile id -> (ti, tj) with ti <= tj over an 8x8 triangle
  int rem = blockIdx.x, ti = 0;
  while (rem >= 8 - ti) { rem -= 8 - ti; ++ti; }
  const int tj   = ti + rem;
  const int head = blockIdx.y;
  const int tid  = threadIdx.x;

  for (int i = tid; i < 16 * HDIM; i += 256) {
    const int r = i / HDIM, e = i % HDIM;
    xa[r][e] = C_EXP2 * Xn[(ti * 16 + r) * DMODEL + head * HDIM + e];
    xb[r][e] = C_EXP2 * Xn[(tj * 16 + r) * DMODEL + head * HDIM + e];
  }
  __syncthreads();

  const int sl = tid >> 4;              // 0..15
  const int tl = tid & 15;              // 0..15

  float a[HDIM];
#pragma unroll
  for (int e = 0; e < HDIM; ++e) a[e] = xa[sl][e];

  float acc0 = 0.f, acc1 = 0.f, acc2 = 0.f, acc3 = 0.f;
  for (int f = 0; f < HDIM; ++f) {
    const float bf = xb[tl][f];
#pragma unroll
    for (int e = 0; e < HDIM; e += 4) {   // 4 accumulators: no serial chain
      float d0 = a[e]     - bf;
      float d1 = a[e + 1] - bf;
      float d2 = a[e + 2] - bf;
      float d3 = a[e + 3] - bf;
      acc0 += __builtin_amdgcn_exp2f(-(d0 * d0));   // v_exp_f32, neg folds
      acc1 += __builtin_amdgcn_exp2f(-(d1 * d1));
      acc2 += __builtin_amdgcn_exp2f(-(d2 * d2));
      acc3 += __builtin_amdgcn_exp2f(-(d3 * d3));
    }
  }
  const float v = (acc0 + acc1 + acc2 + acc3) * (1.0f / HDIM);

  const int s = ti * 16 + sl, t = tj * 16 + tl;
  float* L = attnL + head * SEQ * SEQ;
  L[s * SEQ + t] = v;
  if (ti != tj) L[t * SEQ + s] = v;      // mirror (K is symmetric)
}

// ---------------------------------------------------------------------------
// Kernel 3: row softmax over t, in place on fp32 logits.
// Block = (16-row tile, head); each row handled by 16 threads.
// ---------------------------------------------------------------------------
__global__ __launch_bounds__(256)
void k_softmax(float* __restrict__ attnL) {
  __shared__ float red[16][16];

  const int sTile = blockIdx.x;
  const int head  = blockIdx.y;
  const int tid   = threadIdx.x;
  const int sl = tid >> 4, tc = tid & 15;
  float* row = attnL + (head * SEQ + sTile * 16 + sl) * SEQ;

  float v[8];
#pragma unroll
  for (int i = 0; i < 8; ++i) v[i] = row[tc * 8 + i];

  float pm = -1e30f;
#pragma unroll
  for (int i = 0; i < 8; ++i) pm = fmaxf(pm, v[i]);
  red[sl][tc] = pm;
  __syncthreads();
  float rowmax = -1e30f;
#pragma unroll
  for (int j = 0; j < 16; ++j) rowmax = fmaxf(rowmax, red[sl][j]);
  __syncthreads();

  float ps = 0.f;
#pragma unroll
  for (int i = 0; i < 8; ++i) { v[i] = __expf(v[i] - rowmax); ps += v[i]; }
  red[sl][tc] = ps;
  __syncthreads();
  float tot = 0.f;
#pragma unroll
  for (int j = 0; j < 16; ++j) tot += red[sl][j];
  const float inv = 1.0f / tot;
#pragma unroll
  for (int i = 0; i < 8; ++i) row[tc * 8 + i] = v[i] * inv;
}

// ---------------------------------------------------------------------------
// Kernel 4: out_head = attn @ x_head (128x128 @ 128x32). A-fragment is
// cvt'd fp32->f16 on load (probs <= 1, exactly representable scale).
// ---------------------------------------------------------------------------
__global__ __launch_bounds__(32)
void k_attn_apply(const float* __restrict__ attnL,
                  const _Float16* __restrict__ Xh, _Float16* __restrict__ OH) {
  const int lane  = threadIdx.x;
  const int mTile = blockIdx.x;   // 0..7
  const int nTile = blockIdx.y;   // 0..1
  const int head  = blockIdx.z;   // 0..7
  const int mlo = lane & 15, hi = lane >> 4;
  const float* A = attnL + head * SEQ * SEQ;

  v8f c = {};
  for (int kb = 0; kb < SEQ / 32; ++kb) {
    v16h a, b;
#pragma unroll
    for (int h = 0; h < 16; ++h) {
      const int kA = kb * 32 + hi * 16 + h;
      a[h] = (_Float16)A[(mTile * 16 + mlo) * SEQ + kA];
      const int kB = kb * 32 + hi * 16 + h;   // kB = t index
      b[h] = Xh[kB * DMODEL + head * HDIM + nTile * 16 + mlo];
    }
    c = wmma_f16(a, b, c);
  }
#pragma unroll
  for (int r = 0; r < 8; ++r) {
    const int mg = mTile * 16 + r + 8 * hi;
    OH[mg * DMODEL + head * HDIM + nTile * 16 + mlo] = (_Float16)c[r];
  }
}

// ---------------------------------------------------------------------------
// Kernel 5: final = out_heads @ w_out^T + b_out. f16 WMMA, K=256, fp32 out.
// ---------------------------------------------------------------------------
__global__ __launch_bounds__(32)
void k_out_proj(const _Float16* __restrict__ OH, const float* __restrict__ w_out,
                const float* __restrict__ b_out, float* __restrict__ out) {
  const int lane  = threadIdx.x;
  const int mTile = blockIdx.x;   // 0..7
  const int nTile = blockIdx.y;   // 0..15
  const int mlo = lane & 15, hi = lane >> 4;

  v8f c = {};
  for (int kb = 0; kb < DMODEL / 32; ++kb) {
    v16h a, b;
#pragma unroll
    for (int h = 0; h < 16; ++h) {
      const int kA = kb * 32 + hi * 16 + h;
      a[h] = OH[(mTile * 16 + mlo) * DMODEL + kA];
      const int kB = kb * 32 + hi * 16 + h;
      b[h] = (_Float16)w_out[(nTile * 16 + mlo) * DMODEL + kB];  // B = w_out^T
    }
    c = wmma_f16(a, b, c);
  }
  const float bias = b_out[nTile * 16 + mlo];
#pragma unroll
  for (int r = 0; r < 8; ++r) {
    const int mg = mTile * 16 + r + 8 * hi;
    out[mg * DMODEL + nTile * 16 + mlo] = c[r] + bias;
  }
}

// ---------------------------------------------------------------------------
extern "C" void kernel_launch(void* const* d_in, const int* in_sizes, int n_in,
                              void* d_out, int out_size, void* d_ws, size_t ws_size,
                              hipStream_t stream) {
  const float* inp   = (const float*)d_in[0];
  const float* w_in  = (const float*)d_in[1];
  const float* b_in  = (const float*)d_in[2];
  const float* ln_g  = (const float*)d_in[3];
  const float* ln_b  = (const float*)d_in[4];
  const float* w_out = (const float*)d_in[5];
  const float* b_out = (const float*)d_in[6];
  float* out = (float*)d_out;

  // workspace layout (768 KB total):
  char* ws = (char*)d_ws;
  float*    Xn    = (float*)(ws);                   // 128x256 f32     = 128 KB
  _Float16* Xh    = (_Float16*)(ws + (128u << 10)); // 128x256 f16     =  64 KB
  float*    attnL = (float*)(ws + (192u << 10));    // 8x128x128 f32   = 512 KB
  _Float16* OH    = (_Float16*)(ws + (704u << 10)); // 128x256 f16     =  64 KB

  k_in_proj_ln  <<<dim3(8, 8),    32, 0, stream>>>(inp, w_in, b_in, ln_g, ln_b, Xn, Xh);
  k_gauss_logits<<<dim3(36, 8),  256, 0, stream>>>(Xn, attnL);   // 36 = 8x8 triangle
  k_softmax     <<<dim3(8, 8),   256, 0, stream>>>(attnL);
  k_attn_apply  <<<dim3(8, 2, 8), 32, 0, stream>>>(attnL, Xh, OH);
  k_out_proj    <<<dim3(8, 16),   32, 0, stream>>>(OH, w_out, b_out, out);
}
